// ImgBEVGeneration_9543417332320
// MI455X (gfx1250) — matrix operand order repaired and checked by reading, not compiled, not run on hardware
//
#include <hip/hip_runtime.h>

// ---------------------------------------------------------------------------
// Problem constants (from the reference)
// ---------------------------------------------------------------------------
#define BN_N     6      // B*N cameras
#define DNUM     112    // depth bins
#define FH       16
#define FW       44
#define CIO      80     // channels in/out of conv stack
#define CP       96     // padded channels (multiple of 32 for K-steps)
#define NIMG     96     // BN_N * FH  (conv batch)
#define AH       44     // conv "H"  (= FW)
#define AW       112    // conv "W"  (= D)
#define NWT      7      // W tiles of 16 (112/16)
#define NJOBS    (NIMG * AH * NWT)          // 29568 wave-jobs per conv
#define NPOINTS  (BN_N * DNUM * FH * FW)    // 473088 frustum points
#define NPOS     (NIMG * AH * AW)           // 473088 spatial positions
#define APACK_PER_CONV (5 * 9 * 3 * 32 * 16)  // 69120 bf16 elems per conv
#define NYNX     (128 * 128)

typedef __attribute__((ext_vector_type(16))) __bf16 v16bf;
typedef __attribute__((ext_vector_type(8)))  float  v8f;

union BF16x16 { v16bf v; unsigned short us[16]; };

__device__ __forceinline__ unsigned short f2bf(float f) {
    unsigned u = __float_as_uint(f);
    u += 0x7FFFu + ((u >> 16) & 1u);           // round-to-nearest-even
    return (unsigned short)(u >> 16);
}
__device__ __forceinline__ float bf2f(unsigned short h) {
    return __uint_as_float((unsigned)h << 16);
}

// ---------------------------------------------------------------------------
// Prep: 4x4 inverses + combined projection matrices (6 cameras)
// ---------------------------------------------------------------------------
__device__ void inv4(const float* A, float* out) {
    float a[4][8];
    for (int r = 0; r < 4; ++r)
        for (int c = 0; c < 4; ++c) { a[r][c] = A[r*4 + c]; a[r][4 + c] = (r == c) ? 1.f : 0.f; }
    for (int k = 0; k < 4; ++k) {
        int piv = k; float best = fabsf(a[k][k]);
        for (int r = k + 1; r < 4; ++r) { float v = fabsf(a[r][k]); if (v > best) { best = v; piv = r; } }
        if (piv != k) for (int c = 0; c < 8; ++c) { float t = a[k][c]; a[k][c] = a[piv][c]; a[piv][c] = t; }
        float rinv = 1.0f / a[k][k];
        for (int c = 0; c < 8; ++c) a[k][c] *= rinv;
        for (int r = 0; r < 4; ++r) if (r != k) {
            float f = a[r][k];
            for (int c = 0; c < 8; ++c) a[r][c] -= f * a[k][c];
        }
    }
    for (int r = 0; r < 4; ++r) for (int c = 0; c < 4; ++c) out[r*4 + c] = a[r][4 + c];
}
__device__ void mm4(const float* A, const float* B, float* C) {
    for (int i = 0; i < 4; ++i)
        for (int j = 0; j < 4; ++j) {
            float s = 0.f;
            for (int k = 0; k < 4; ++k) s += A[i*4 + k] * B[k*4 + j];
            C[i*4 + j] = s;
        }
}

__global__ void prep_mats_kernel(const float* __restrict__ s2e, const float* __restrict__ intrin,
                                 const float* __restrict__ ida, const float* __restrict__ bda,
                                 float* __restrict__ M1, float* __restrict__ M2) {
    int n = threadIdx.x;
    if (n >= BN_N) return;
    inv4(ida + n*16, M1 + n*16);               // ida^-1
    float iinv[16], t[16];
    inv4(intrin + n*16, iinv);                 // intrin^-1
    mm4(s2e + n*16, iinv, t);                  // s2e @ intrin^-1
    mm4(bda, t, M2 + n*16);                    // bda @ s2e @ intrin^-1
}

// ---------------------------------------------------------------------------
// Prep: fold BN params -> scale/shift  (3 BN sets of 80)
// ---------------------------------------------------------------------------
__global__ void prep_bn_kernel(const float* __restrict__ bn0, const float* __restrict__ bn1,
                               const float* __restrict__ bn2,
                               float* __restrict__ scale, float* __restrict__ shift) {
    int t = threadIdx.x;
    if (t >= 3 * CIO) return;
    int set = t / CIO, c = t % CIO;
    const float* p = (set == 0) ? bn0 : (set == 1) ? bn1 : bn2;
    float g = p[c], b = p[CIO + c], m = p[2*CIO + c], v = p[3*CIO + c];
    float s = g * rsqrtf(v + 1e-5f);
    scale[t] = s;
    shift[t] = b - m * s;
}

// ---------------------------------------------------------------------------
// Prep: pack OIHW f32 weights into bf16 WMMA A-fragments (16x32, bf16 layout)
//   index = ((((conv*5+mt)*9+tap)*3+ks)*32+lane)*16 + e
//   lane<16 : K=e     (0..15)
//   lane>=16: K=16+e  (16..31)
// ---------------------------------------------------------------------------
__global__ __launch_bounds__(256)
void pack_w_kernel(const float* __restrict__ w0, const float* __restrict__ w1,
                   const float* __restrict__ w2, const float* __restrict__ w3,
                   unsigned short* __restrict__ apack) {
    int t = blockIdx.x * 256 + threadIdx.x;        // < 4*69120 = 276480 (exact grid)
    int e    = t & 15;
    int r1   = t >> 4;
    int lane = r1 & 31;  r1 >>= 5;
    int ks   = r1 % 3;   r1 /= 3;
    int tap  = r1 % 9;   r1 /= 9;
    int mt   = r1 % 5;
    int conv = r1 / 5;
    const float* w = (conv == 0) ? w0 : (conv == 1) ? w1 : (conv == 2) ? w2 : w3;
    int klo = (lane < 16) ? e
                          : (16 + e);
    int cin = ks * 32 + klo;
    int m   = mt * 16 + (lane & 15);
    float val = (cin < CIO) ? w[(m * CIO + cin) * 9 + tap] : 0.0f;
    apack[t] = f2bf(val);
}

// ---------------------------------------------------------------------------
// Geometry: per frustum point -> voxel index (-1 = invalid)
//   point layout p = ((bn*112 + d)*16 + fh)*44 + fw
//   NOTE: validity is tested on the TRUNCATED indices only, exactly like the
//   reference's astype(int32) + integer range check (fractional indices in
//   (-1,0) truncate to 0 and count as valid).
// ---------------------------------------------------------------------------
__global__ __launch_bounds__(256)
void geom_kernel(const float* __restrict__ M1, const float* __restrict__ M2,
                 int* __restrict__ vidx) {
    int t = blockIdx.x * 256 + threadIdx.x;        // < 473088 (exact grid)
    int fw = t % FW;  int r = t / FW;
    int fh = r % FH;  r /= FH;
    int d  = r % DNUM;
    int bn = r / DNUM;
    float xs = fw * (703.0f / 43.0f);              // linspace(0,703,44)
    float ys = fh * 17.0f;                          // linspace(0,255,16)
    float dc = 2.0f + 0.5f * (float)d;              // arange(2,58,0.5)
    const float* m1 = M1 + bn * 16;
    const float* m2 = M2 + bn * 16;
    float u0 = m1[0]*xs + m1[1]*ys + m1[2]*dc + m1[3];
    float u1 = m1[4]*xs + m1[5]*ys + m1[6]*dc + m1[7];
    float u2 = m1[8]*xs + m1[9]*ys + m1[10]*dc + m1[11];
    float u3 = m1[12]*xs + m1[13]*ys + m1[14]*dc + m1[15];
    float p0 = u0 * u2, p1 = u1 * u2, p2 = u2, p3 = u3;
    float gx = m2[0]*p0 + m2[1]*p1 + m2[2]*p2 + m2[3]*p3;
    float gy = m2[4]*p0 + m2[5]*p1 + m2[6]*p2 + m2[7]*p3;
    float gz = m2[8]*p0 + m2[9]*p1 + m2[10]*p2 + m2[11]*p3;
    int ix = (int)((gx + 51.2f) / 0.8f);           // trunc toward zero == astype(int32)
    int iy = (int)((gy + 51.2f) / 0.8f);
    int iz = (int)((gz + 5.0f) / 8.0f);
    bool valid = (ix >= 0) && (ix < 128) &&
                 (iy >= 0) && (iy < 128) &&
                 (iz == 0);
    vidx[t] = valid ? (iy * 128 + ix) : -1;
}

// ---------------------------------------------------------------------------
// Fuse: act0[pos][c] = depth[bn,d,fh,fw] * img[bn,c,fh,fw]  (bf16, channel-last)
//   pos = ((img*44 + fw)*112 + d),  img = bn*16 + fh
// ---------------------------------------------------------------------------
__global__ __launch_bounds__(256)
void fuse_kernel(const float* __restrict__ imgf, const float* __restrict__ dep,
                 unsigned short* __restrict__ act0) {
    unsigned t = blockIdx.x * 256u + threadIdx.x;  // < 473088*96 (exact grid)
    int c = (int)(t % CP);
    unsigned pos = t / CP;
    int d  = (int)(pos % AW);  unsigned r = pos / AW;
    int fw = (int)(r % AH);
    int ni = (int)(r / AH);
    int bn = ni >> 4, fh = ni & 15;
    float v = 0.0f;
    if (c < CIO) {
        float iv = imgf[((bn * CIO + c) * FH + fh) * FW + fw];
        float dv = dep[((bn * DNUM + d) * FH + fh) * FW + fw];
        v = iv * dv;
    }
    act0[(size_t)pos * CP + c] = f2bf(v);
}

// ---------------------------------------------------------------------------
// Implicit-GEMM 3x3 conv via v_wmma_f32_16x16x32_bf16.
// One wave per (img, row, 16-col tile); 5 M-tiles (80 out channels).
// mode 0: out = relu(bn(conv))                       -> bf16 actout
// mode 1: out = relu(bn(conv)) + resid               -> bf16 actout
// mode 2: out = conv + bias, voxel-scatter atomicAdd -> f32 d_out
// ---------------------------------------------------------------------------
__global__ __launch_bounds__(256)
void conv_wmma_kernel(const unsigned short* __restrict__ actin,
                      const unsigned short* __restrict__ apack,
                      const float* __restrict__ scale,
                      const float* __restrict__ shift,
                      const unsigned short* __restrict__ resid,
                      unsigned short* __restrict__ actout,
                      const int* __restrict__ vidx,
                      const float* __restrict__ bias,
                      float* __restrict__ out,
                      int mode) {
    int wave = threadIdx.x >> 5;
    int lane = threadIdx.x & 31;
    int job  = blockIdx.x * 8 + wave;
    if (job >= NJOBS) return;                      // uniform per wave (exact grid anyway)
    int wt  = job % NWT;  int t = job / NWT;
    int h   = t % AH;
    int img = t / AH;

    v8f acc[5] = {};
    int ncol  = lane & 15;
    int ow    = wt * 16 + ncol;
    int bhalf = (lane < 16) ? 0 : 16;              // B frag: contiguous K run per lane

    for (int tap = 0; tap < 9; ++tap) {
        int r = tap / 3, s = tap % 3;
        int ih = h + r - 1;
        int iw = ow + s - 1;
        bool inb = (ih >= 0) && (ih < AH) && (iw >= 0) && (iw < AW);
        const unsigned short* prow =
            actin + (((size_t)img * AH + (inb ? ih : 0)) * AW + (inb ? iw : 0)) * CP;
        const unsigned short* ap0 = apack + (size_t)(tap * 3) * 32 * 16;
#pragma unroll
        for (int ks = 0; ks < 3; ++ks) {
            BF16x16 b;
            if (inb) {
                b.v = *(const v16bf*)(const void*)(prow + ks * 32 + bhalf);
            } else {
#pragma unroll
                for (int i = 0; i < 16; ++i) b.us[i] = 0;
            }
            const unsigned short* ap = ap0 + (ks * 32 + lane) * 16;
#pragma unroll
            for (int m = 0; m < 5; ++m) {
                BF16x16 a;
                a.v = *(const v16bf*)(const void*)(ap + (size_t)m * (9 * 3 * 32 * 16));
                acc[m] = __builtin_amdgcn_wmma_f32_16x16x32_bf16(
                    false, a.v, false, b.v, (short)0, acc[m], false, false);
            }
        }
    }

    // C layout: VGPR rr -> M = rr (+8 for lanes 16-31), N = lane&15
    int moff = (lane >> 4) * 8;
    size_t pos = ((size_t)img * AH + h) * AW + ow;
    if (mode == 2) {
        int bn = img >> 4, fh = img & 15;
        int p = ((bn * AW + ow) * FH + fh) * FW + h;   // (bn, d=ow, fh, fw=h)
        int vid = vidx[p];
#pragma unroll
        for (int m = 0; m < 5; ++m)
#pragma unroll
            for (int rr = 0; rr < 8; ++rr) {
                int cout = m * 16 + rr + moff;
                float v = acc[m][rr] + bias[cout];
                if (vid >= 0) atomicAdd(out + (size_t)cout * NYNX + vid, v);
            }
    } else {
#pragma unroll
        for (int m = 0; m < 5; ++m)
#pragma unroll
            for (int rr = 0; rr < 8; ++rr) {
                int cout = m * 16 + rr + moff;
                float v = acc[m][rr] * scale[cout] + shift[cout];
                v = fmaxf(v, 0.0f);
                if (mode == 1) v += bf2f(resid[pos * CP + cout]);
                actout[pos * CP + cout] = f2bf(v);
            }
    }
}

// ---------------------------------------------------------------------------
// Host launcher
// ---------------------------------------------------------------------------
extern "C" void kernel_launch(void* const* d_in, const int* in_sizes, int n_in,
                              void* d_out, int out_size, void* d_ws, size_t ws_size,
                              hipStream_t stream) {
    (void)in_sizes; (void)n_in; (void)ws_size;
    const float* img_feats = (const float*)d_in[0];
    const float* depth     = (const float*)d_in[1];
    const float* s2e       = (const float*)d_in[2];
    const float* intrin    = (const float*)d_in[3];
    const float* ida       = (const float*)d_in[4];
    const float* bda       = (const float*)d_in[5];
    const float* w_red     = (const float*)d_in[6];
    const float* bn_red    = (const float*)d_in[7];
    const float* w_c1      = (const float*)d_in[8];
    const float* bn_c1     = (const float*)d_in[9];
    const float* w_c2      = (const float*)d_in[10];
    const float* bn_c2     = (const float*)d_in[11];
    const float* w_out     = (const float*)d_in[12];
    const float* b_out     = (const float*)d_in[13];
    float* out = (float*)d_out;

    char* ws = (char*)d_ws;
    const size_t SZ_ACT = (size_t)NPOS * CP * sizeof(unsigned short);   // 90,832,896 B
    unsigned short* bufA  = (unsigned short*)(ws);
    unsigned short* bufB  = (unsigned short*)(ws + SZ_ACT);
    unsigned short* bufC  = (unsigned short*)(ws + 2 * SZ_ACT);
    unsigned short* apack = (unsigned short*)(ws + 3 * SZ_ACT);
    const size_t SZ_APACK = (size_t)4 * APACK_PER_CONV * sizeof(unsigned short);
    int*   vidx  = (int*)  (ws + 3 * SZ_ACT + SZ_APACK);
    float* M1    = (float*)(ws + 3 * SZ_ACT + SZ_APACK + (size_t)NPOINTS * sizeof(int));
    float* M2    = M1 + BN_N * 16;
    float* scale = M2 + BN_N * 16;
    float* shift = scale + 3 * CIO;

    // zero output + pad channels of ping-pong activation buffers
    hipMemsetAsync(out, 0, (size_t)out_size * sizeof(float), stream);
    hipMemsetAsync(bufB, 0, SZ_ACT, stream);
    hipMemsetAsync(bufC, 0, SZ_ACT, stream);

    prep_mats_kernel<<<1, 64, 0, stream>>>(s2e, intrin, ida, bda, M1, M2);
    prep_bn_kernel<<<1, 256, 0, stream>>>(bn_red, bn_c1, bn_c2, scale, shift);
    pack_w_kernel<<<(4 * APACK_PER_CONV) / 256, 256, 0, stream>>>(w_red, w_c1, w_c2, w_out, apack);
    geom_kernel<<<NPOINTS / 256, 256, 0, stream>>>(M1, M2, vidx);
    fuse_kernel<<<((unsigned)NPOS * CP) / 256, 256, 0, stream>>>(img_feats, depth, bufA);

    const int CG = NJOBS / 8;   // 3696 blocks, 8 waves each
    // layer 1: act1 = relu(bn_red(conv(act0, w_red)))
    conv_wmma_kernel<<<CG, 256, 0, stream>>>(bufA, apack + 0 * APACK_PER_CONV,
        scale + 0, shift + 0, nullptr, bufB, nullptr, nullptr, nullptr, 0);
    // layer 2: act2 = relu(bn_c1(conv(act1, w_c1)))
    conv_wmma_kernel<<<CG, 256, 0, stream>>>(bufB, apack + 1 * APACK_PER_CONV,
        scale + CIO, shift + CIO, nullptr, bufA, nullptr, nullptr, nullptr, 0);
    // layer 3: act3 = relu(bn_c2(conv(act2, w_c2))) + act1   (residual)
    conv_wmma_kernel<<<CG, 256, 0, stream>>>(bufA, apack + 2 * APACK_PER_CONV,
        scale + 2 * CIO, shift + 2 * CIO, bufB, bufC, nullptr, nullptr, nullptr, 1);
    // layer 4: conv(act3, w_out) + b_out, voxel-pool scatter into output
    conv_wmma_kernel<<<CG, 256, 0, stream>>>(bufC, apack + 3 * APACK_PER_CONV,
        nullptr, nullptr, nullptr, nullptr, vidx, b_out, out, 2);
}
